// GaussianMixture_2783138808281
// MI455X (gfx1250) — matrix operand order repaired and checked by reading, not compiled, not run on hardware
//
#include <hip/hip_runtime.h>

// ---------------------------------------------------------------------------
// GMM noise-conditioned score, N=16384, K=128, D=32 on gfx1250.
// Per 16-sample tile and per k:
//   U^T = P(Q^T) * Y^T        (2x v_wmma_f32_16x16x32_f16, K-dim = D = 32)
//   T^T = U^T / (L_eig+sig^2) (elementwise, in WMMA C/D layout)
//   Z^T = Q * T^T             (2x v_wmma_f32_16x16x32_f16)
//   w   = phi * rsqrt((2pi)^32 * prod L) * exp(-quad/2); num += w*Z; den += w
// P(.) is a row permutation applied when pre-swizzling Q^T to f16: A-tile t
// row m holds Q^T[16*(m>>3) + 8t + (m&7)], so after matmul1 each lane group g
// holds exactly j in [16g,16g+16) -- the K-rows the second matmul's B operand
// needs from that lane (in-lane repack, no shuffles). Y = mu - x is computed
// in packed f16 (v_pk_sub_f16) from f16-staged mu (LDS) and f16 x (regs).
// ---------------------------------------------------------------------------

typedef __attribute__((ext_vector_type(16))) _Float16 v16h;
typedef __attribute__((ext_vector_type(8)))  float    v8f;

union V16H {
    v16h     v;
    _Float16 e[16];
    uint4    q[2];
};

#define NSAMP 16384
#define KMIX  128
#define DIM   32
#define TWO_PI_POW_D 3.4820437e25f    // (2*pi)^32
#define NEG_HALF_LOG2E (-0.72134752f) // -0.5 * log2(e)

// ---- pre-pass: Q (f32 [K][32][32]) -> f16 Q and f16 row-permuted Q^T ------
__global__ __launch_bounds__(256) void gmm_prep_q(const float* __restrict__ Q,
                                                  _Float16* __restrict__ Qh,
                                                  _Float16* __restrict__ QTh) {
    int idx = blockIdx.x * 256 + threadIdx.x;  // K*32*32 = 131072 elements
    int k   = idx >> 10;
    int row = (idx >> 5) & 31;   // l
    int col = idx & 31;          // j
    float v = Q[idx];
    Qh[idx] = (_Float16)v;       // Q[k][row][col], natural order (matmul2 A)
    // permuted Q^T: buffer row p holds Q^T[j]; p = 16*((j>>3)&1) + 8*(j>>4) + (j&7)
    int p = (((col >> 3) & 1) << 4) + ((col >> 4) << 3) + (col & 7);
    QTh[(k << 10) + (p << 5) + row] = (_Float16)v;
}

// ---- main kernel ----------------------------------------------------------
__global__ __launch_bounds__(256) void gmm_score(const float* __restrict__ x,
                                                 const float* __restrict__ sigma,
                                                 const float* __restrict__ phi,
                                                 const float* __restrict__ mu,
                                                 const float* __restrict__ le,
                                                 const _Float16* __restrict__ Qh,
                                                 const _Float16* __restrict__ QTh,
                                                 float* __restrict__ out) {
    __shared__ __align__(16) float     le_s[KMIX * DIM];   // 16 KB f32
    __shared__ __align__(16) _Float16  muh_s[KMIX * DIM];  //  8 KB f16

    const int tid = threadIdx.x;
    {   // cooperative LDS stage: L_eig f32, mu converted to f16
        const float4* l4 = (const float4*)le;
        float4* ls = (float4*)le_s;
#pragma unroll
        for (int j = 0; j < 4; ++j) ls[tid + j * 256] = l4[tid + j * 256];

        const float4* m4 = (const float4*)(mu + tid * 16);
        union { _Float16 h[16]; uint4 q[2]; } t16;
#pragma unroll
        for (int i = 0; i < 4; ++i) {
            float4 v = m4[i];
            t16.h[4*i+0] = (_Float16)v.x; t16.h[4*i+1] = (_Float16)v.y;
            t16.h[4*i+2] = (_Float16)v.z; t16.h[4*i+3] = (_Float16)v.w;
        }
        uint4* mh4 = (uint4*)muh_s;
        mh4[tid * 2]     = t16.q[0];
        mh4[tid * 2 + 1] = t16.q[1];
    }
    __syncthreads();

    const int wave = tid >> 5;
    const int lane = tid & 31;
    const int g    = lane >> 4;   // lane group
    const int c    = lane & 15;   // column -> sample index
    const int n0   = blockIdx.x * 128 + wave * 16;
    const int n    = n0 + c;

    const float sg   = sigma[n];
    const float sig2 = sg * sg;

    // x slice (cols 16g..16g+15) converted once to packed f16
    V16H xh;
    {
        const float4* xp = (const float4*)(x + n * DIM + 16 * g);
#pragma unroll
        for (int i = 0; i < 4; ++i) {
            float4 v = xp[i];
            xh.e[4*i+0] = (_Float16)v.x; xh.e[4*i+1] = (_Float16)v.y;
            xh.e[4*i+2] = (_Float16)v.z; xh.e[4*i+3] = (_Float16)v.w;
        }
    }

    float num0[8], num1[8];
#pragma unroll
    for (int r = 0; r < 8; ++r) { num0[r] = 0.0f; num1[r] = 0.0f; }
    float den = 0.0f;

    const v8f zero8 = 0.0f;

    for (int k = 0; k < KMIX; ++k) {
        const _Float16* qtk = QTh + (k << 10);
        const _Float16* qk  = Qh  + (k << 10);

        // A operands: two contiguous b128 per 16x16 tile (pre-swizzled f16)
        V16H a1_0, a1_1, a2_0, a2_1;
        a1_0.q[0] = *(const uint4*)(qtk + (c << 5)              + 16 * g);
        a1_0.q[1] = *(const uint4*)(qtk + (c << 5)         + 8 + 16 * g);
        a1_1.q[0] = *(const uint4*)(qtk + ((16 + c) << 5)       + 16 * g);
        a1_1.q[1] = *(const uint4*)(qtk + ((16 + c) << 5)  + 8 + 16 * g);
        a2_0.q[0] = *(const uint4*)(qk  + (c << 5)              + 16 * g);
        a2_0.q[1] = *(const uint4*)(qk  + (c << 5)         + 8 + 16 * g);
        a2_1.q[0] = *(const uint4*)(qk  + ((16 + c) << 5)       + 16 * g);
        a2_1.q[1] = *(const uint4*)(qk  + ((16 + c) << 5)  + 8 + 16 * g);

        // B operand: Y^T = mu - x in packed f16 (lane = column n=c, rows l=i+16g)
        V16H mh, by;
        mh.q[0] = *(const uint4*)(muh_s + k * DIM + 16 * g);
        mh.q[1] = *(const uint4*)(muh_s + k * DIM + 16 * g + 8);
        by.v = mh.v - xh.v;   // 8x v_pk_sub_f16

        // matmul1: permuted U^T; lane(g,c) tile t reg r = U^T[16g+8t+r, n0+c]
        v8f u0 = __builtin_amdgcn_wmma_f32_16x16x32_f16(false, a1_0.v, false, by.v,
                                                        (short)0, zero8, false, false);
        v8f u1 = __builtin_amdgcn_wmma_f32_16x16x32_f16(false, a1_1.v, false, by.v,
                                                        (short)0, zero8, false, false);

        // L slice for this lane: j = 16g + 0..15 (contiguous)
        float lv[16];
        {
            const float4* lp = (const float4*)(le_s + k * DIM + 16 * g);
            float4 a = lp[0], b = lp[1], cc = lp[2], d = lp[3];
            lv[0]=a.x;  lv[1]=a.y;  lv[2]=a.z;  lv[3]=a.w;
            lv[4]=b.x;  lv[5]=b.y;  lv[6]=b.z;  lv[7]=b.w;
            lv[8]=cc.x; lv[9]=cc.y; lv[10]=cc.z;lv[11]=cc.w;
            lv[12]=d.x; lv[13]=d.y; lv[14]=d.z; lv[15]=d.w;
        }

        // elementwise scale + per-r partial products (independent, tree-reduced)
        float ts0[8], ts1[8], pr[8], qr[8];
#pragma unroll
        for (int r = 0; r < 8; ++r) {
            float L0  = lv[r] + sig2;        // j = 16g + r      (tile 0)
            float L1  = lv[8 + r] + sig2;    // j = 16g + 8 + r  (tile 1)
            float ri0 = __builtin_amdgcn_rcpf(L0);
            float ri1 = __builtin_amdgcn_rcpf(L1);
            float t0  = u0[r] * ri0;
            float t1  = u1[r] * ri1;
            pr[r] = L0 * L1;
            qr[r] = u0[r] * t0 + u1[r] * t1;
            ts0[r] = t0; ts1[r] = t1;
        }
        float p01 = pr[0] * pr[1], p23 = pr[2] * pr[3];
        float p45 = pr[4] * pr[5], p67 = pr[6] * pr[7];
        float prod = (p01 * p23) * (p45 * p67);
        float q01 = qr[0] + qr[1], q23 = qr[2] + qr[3];
        float q45 = qr[4] + qr[5], q67 = qr[6] + qr[7];
        float quad = (q01 + q23) + (q45 + q67);

        // combine with partner lane (other 16 j's, same sample n)
        quad += __shfl_xor(quad, 16);
        prod *= __shfl_xor(prod, 16);

        // phi[k] is loop-uniform -> scalar (SMEM) load path
        float w = phi[k] * __builtin_amdgcn_rsqf(TWO_PI_POW_D * prod) *
                  __builtin_amdgcn_exp2f(NEG_HALF_LOG2E * quad);
        den += w;

        // B operand T^T: half i -> K row j = i + 16g; in-lane repack
        V16H b2;
#pragma unroll
        for (int i = 0; i < 8; ++i) {
            b2.e[i]     = (_Float16)ts0[i];
            b2.e[i + 8] = (_Float16)ts1[i];
        }

        // matmul2: Z^T[i, n] tiles (natural row order)
        v8f z0 = __builtin_amdgcn_wmma_f32_16x16x32_f16(false, a2_0.v, false, b2.v,
                                                        (short)0, zero8, false, false);
        v8f z1 = __builtin_amdgcn_wmma_f32_16x16x32_f16(false, a2_1.v, false, b2.v,
                                                        (short)0, zero8, false, false);
#pragma unroll
        for (int r = 0; r < 8; ++r) {
            num0[r] += w * z0[r];
            num1[r] += w * z1[r];
        }
    }

    // epilogue: lane(g,c) holds Z rows i = r+8g (tile0) and 16+r+8g (tile1)
    float invden = __builtin_amdgcn_rcpf(den);
    float* orow = out + n * DIM;
    ((float4*)(orow + 8 * g))[0] =
        make_float4(num0[0]*invden, num0[1]*invden, num0[2]*invden, num0[3]*invden);
    ((float4*)(orow + 8 * g))[1] =
        make_float4(num0[4]*invden, num0[5]*invden, num0[6]*invden, num0[7]*invden);
    ((float4*)(orow + 16 + 8 * g))[0] =
        make_float4(num1[0]*invden, num1[1]*invden, num1[2]*invden, num1[3]*invden);
    ((float4*)(orow + 16 + 8 * g))[1] =
        make_float4(num1[4]*invden, num1[5]*invden, num1[6]*invden, num1[7]*invden);
}

extern "C" void kernel_launch(void* const* d_in, const int* in_sizes, int n_in,
                              void* d_out, int out_size, void* d_ws, size_t ws_size,
                              hipStream_t stream) {
    (void)in_sizes; (void)n_in; (void)out_size; (void)ws_size;
    const float* x     = (const float*)d_in[0];
    const float* sigma = (const float*)d_in[1];
    const float* phi   = (const float*)d_in[2];
    const float* mu    = (const float*)d_in[3];
    const float* le    = (const float*)d_in[4];
    const float* Q     = (const float*)d_in[5];
    float* out = (float*)d_out;

    // d_ws layout: [0, 256KB) f16 Q, [256KB, 512KB) f16 permuted Q^T
    _Float16* Qh  = (_Float16*)d_ws;
    _Float16* QTh = (_Float16*)((char*)d_ws + (size_t)KMIX * DIM * DIM * sizeof(_Float16));

    hipLaunchKernelGGL(gmm_prep_q, dim3((KMIX * DIM * DIM) / 256), dim3(256), 0, stream,
                       Q, Qh, QTh);
    hipLaunchKernelGGL(gmm_score, dim3(NSAMP / 128), dim3(256), 0, stream,
                       x, sigma, phi, mu, le, Qh, QTh, out);
}
